// MultiheadAttentionBlock_47356309406430
// MI455X (gfx1250) — matrix operand order repaired and checked by reading, not compiled, not run on hardware
//
#include <hip/hip_runtime.h>

typedef __attribute__((ext_vector_type(16))) _Float16 v16h;
typedef __attribute__((ext_vector_type(8)))  _Float16 v8h;
typedef __attribute__((ext_vector_type(2)))  __fp16   v2hf;  // cvt_pkrtz result type
typedef __attribute__((ext_vector_type(8)))  float    v8f;

#define D_MODEL 1024
#define NH      16
#define DKH     64
#define BATCH   8
#define SEQT    1024
#define MROWS   (BATCH * SEQT)   // 8192

union Frag {
    v16h v;
    v8h  h[2];
    v2hf pk[8];   // element pair (2k, 2k+1) of v, as __fp16x2
};

static __device__ __forceinline__ v8f wmma16(v16h a, v16h b, v8f c) {
    // D = A(16x32 f16) * B(32x16 f16) + C(16x16 f32)
    return __builtin_amdgcn_wmma_f32_16x16x32_f16(
        /*neg_a=*/false, a, /*neg_b=*/false, b,
        /*c_mod=*/(short)0, c, /*reuse_a=*/false, /*reuse_b=*/false);
}

// ---------------------------------------------------------------------------
// Stage 0: f32 -> f16 conversion (8 elements / thread)
// ---------------------------------------------------------------------------
__global__ void cvt_f32_f16(const float* __restrict__ in,
                            _Float16* __restrict__ out, int n) {
    int i = (blockIdx.x * blockDim.x + threadIdx.x) * 8;
    if (i + 8 <= n) {
        float4 a = *(const float4*)(in + i);
        float4 b = *(const float4*)(in + i + 4);
        Frag o;
        o.pk[0] = __builtin_amdgcn_cvt_pkrtz(a.x, a.y);
        o.pk[1] = __builtin_amdgcn_cvt_pkrtz(a.z, a.w);
        o.pk[2] = __builtin_amdgcn_cvt_pkrtz(b.x, b.y);
        o.pk[3] = __builtin_amdgcn_cvt_pkrtz(b.z, b.w);
        *(v8h*)(out + i) = o.h[0];
    }
}

// ---------------------------------------------------------------------------
// Stage 1/3: C[M,N] = A[M,K] * W[N,K]^T + bias
//   OUT_MODE 0: f16 row-major [M, N]            (Q/K projections, x)
//   OUT_MODE 1: f32 row-major [M, N]            (final output)
//   OUT_MODE 2: f16 transposed head-split V^T [B, H, DK, T]  (V projection)
// Block: 256 threads = 8 waves (2x4), wave tile 64x64, block tile 128x256.
// ---------------------------------------------------------------------------
template <int OUT_MODE>
__global__ __launch_bounds__(256)
void gemm_wmma(const _Float16* __restrict__ A, const _Float16* __restrict__ W,
               const float* __restrict__ bias, _Float16* __restrict__ outH,
               float* __restrict__ outF, int M, int N, int K) {
    const int lane = threadIdx.x & 31;
    const int ln   = lane & 15;
    const int hi   = lane >> 4;
    const int wave = threadIdx.x >> 5;     // 0..7
    const int wm   = wave >> 2;            // 0..1
    const int wn   = wave & 3;             // 0..3
    const int nBN  = N >> 8;               // blocks along N (256 each)
    const int bm   = blockIdx.x / nBN;
    const int bn   = blockIdx.x % nBN;
    const int row0 = bm * 128 + wm * 64;
    const int col0 = bn * 256 + wn * 64;

    v8f acc[4][4];
#pragma unroll
    for (int i = 0; i < 4; i++)
#pragma unroll
        for (int j = 0; j < 4; j++) acc[i][j] = (v8f)(0.0f);

    for (int k0 = 0; k0 < K; k0 += 32) {
        Frag a[4], b[4];
#pragma unroll
        for (int mi = 0; mi < 4; mi++) {
            // A fragment: lane ln = row, K chunks at +0 and +16 (doc layout)
            const _Float16* pa =
                A + (size_t)(row0 + mi * 16 + ln) * K + k0 + hi * 8;
            a[mi].h[0] = *(const v8h*)pa;
            a[mi].h[1] = *(const v8h*)(pa + 16);
        }
#pragma unroll
        for (int ni = 0; ni < 4; ni++) {
            // B fragment: lane ln = output column = W row; K chunks at +0 and +16
            const _Float16* pb =
                W + (size_t)(col0 + ni * 16 + ln) * K + k0 + hi * 8;
            b[ni].h[0] = *(const v8h*)pb;
            b[ni].h[1] = *(const v8h*)(pb + 16);
        }
        if (k0 + 32 < K)
            __builtin_prefetch(A + (size_t)(row0 + ln) * K + k0 + 32, 0, 1);
#pragma unroll
        for (int mi = 0; mi < 4; mi++)
#pragma unroll
            for (int ni = 0; ni < 4; ni++)
                acc[mi][ni] = wmma16(a[mi].v, b[ni].v, acc[mi][ni]);
    }

#pragma unroll
    for (int mi = 0; mi < 4; mi++) {
#pragma unroll
        for (int ni = 0; ni < 4; ni++) {
            const int col = col0 + ni * 16 + ln;
            const float bv = bias[col];
#pragma unroll
            for (int r = 0; r < 8; r++) {
                const int m = row0 + mi * 16 + r + hi * 8;
                const float val = acc[mi][ni][r] + bv;
                if (OUT_MODE == 0) {
                    outH[(size_t)m * N + col] = (_Float16)val;
                } else if (OUT_MODE == 1) {
                    outF[(size_t)m * N + col] = val;
                } else {
                    // V^T head-split: [B, H, DK, T]
                    const int bidx = m >> 10;        // batch
                    const int t    = m & 1023;       // time
                    const int hh   = col >> 6;       // head
                    const int d    = col & 63;       // dim in head
                    outH[((size_t)(bidx * NH + hh) * DKH + d) * SEQT + t] =
                        (_Float16)val;
                }
            }
        }
    }
}

// ---------------------------------------------------------------------------
// Stage 2: attention. One wave per (b, h, 16-query block).
// S^T = K * Q^T via WMMA (C layout of S^T == A layout of P), online softmax,
// O += P * V with V read from V^T, final 1/l normalize, write x f16 [B*T, D].
// ---------------------------------------------------------------------------
__global__ __launch_bounds__(256)
void attn_kernel(const _Float16* __restrict__ Qp, const _Float16* __restrict__ Kp,
                 const _Float16* __restrict__ Vt, const int* __restrict__ mask,
                 _Float16* __restrict__ X) {
    const int lane = threadIdx.x & 31;
    const int ln   = lane & 15;
    const int hi   = lane >> 4;
    const int gw   = (int)((blockIdx.x * blockDim.x + threadIdx.x) >> 5);
    const int qb   = gw & 63;            // T/16 query blocks
    const int hd   = (gw >> 6) & 15;     // head
    const int bb   = gw >> 10;           // batch
    const int q0   = qb * 16;

    // Q/K are row-major [B*T, D_MODEL]; head slice is contiguous within a row
    const _Float16* Qh = Qp + (size_t)bb * SEQT * D_MODEL + hd * DKH;
    const _Float16* Kh = Kp + (size_t)bb * SEQT * D_MODEL + hd * DKH;
    // V^T is [B, H, DK, T]
    const _Float16* Vh = Vt + (size_t)(bb * NH + hd) * DKH * SEQT;
    const int* Mrow = mask + (size_t)bb * SEQT * SEQT + (size_t)(q0 + ln) * SEQT;

    // Q^T B-fragments (K-dim = d, N = query i = ln), invariant over key loop
    Frag qf[2];
#pragma unroll
    for (int t = 0; t < 2; t++) {
        const _Float16* pq =
            Qh + (size_t)(q0 + ln) * D_MODEL + t * 32 + hi * 8;
        qf[t].h[0] = *(const v8h*)pq;
        qf[t].h[1] = *(const v8h*)(pq + 16);
    }

    v8f o[4];
#pragma unroll
    for (int nt = 0; nt < 4; nt++) o[nt] = (v8f)(0.0f);
    float mrun = -1e30f, lrun = 0.0f;

    for (int kb = 0; kb < SEQT; kb += 32) {
        // K A-fragments: 2 key tiles (j) x 2 d-chunks
        Frag kf[2][2];
#pragma unroll
        for (int jt = 0; jt < 2; jt++)
#pragma unroll
            for (int t = 0; t < 2; t++) {
                const _Float16* pk =
                    Kh + (size_t)(kb + jt * 16 + ln) * D_MODEL + t * 32 + hi * 8;
                kf[jt][t].h[0] = *(const v8h*)pk;
                kf[jt][t].h[1] = *(const v8h*)(pk + 16);
            }

        v8f s0 = (v8f)(0.0f), s1 = (v8f)(0.0f);
        s0 = wmma16(kf[0][0].v, qf[0].v, s0);
        s0 = wmma16(kf[0][1].v, qf[1].v, s0);
        s1 = wmma16(kf[1][0].v, qf[0].v, s1);
        s1 = wmma16(kf[1][1].v, qf[1].v, s1);

        // mask: for lane = query row q0+ln, 8 consecutive j per half-wave
        int mk0[8], mk1[8];
        {
            const int4* p0 = (const int4*)(Mrow + kb + hi * 8);
            int4 x0 = p0[0], x1 = p0[1];
            mk0[0] = x0.x; mk0[1] = x0.y; mk0[2] = x0.z; mk0[3] = x0.w;
            mk0[4] = x1.x; mk0[5] = x1.y; mk0[6] = x1.z; mk0[7] = x1.w;
            const int4* p1 = (const int4*)(Mrow + kb + 16 + hi * 8);
            int4 y0 = p1[0], y1 = p1[1];
            mk1[0] = y0.x; mk1[1] = y0.y; mk1[2] = y0.z; mk1[3] = y0.w;
            mk1[4] = y1.x; mk1[5] = y1.y; mk1[6] = y1.z; mk1[7] = y1.w;
        }
        float sv0[8], sv1[8];
#pragma unroll
        for (int r = 0; r < 8; r++) {
            sv0[r] = (mk0[r] == 0) ? 1e-9f : s0[r] * 0.125f;  // 1/sqrt(64)
            sv1[r] = (mk1[r] == 0) ? 1e-9f : s1[r] * 0.125f;
        }

        // online softmax: row (query) stats live in lane ln and ln+16
        float rm = -1e30f;
#pragma unroll
        for (int r = 0; r < 8; r++) rm = fmaxf(rm, fmaxf(sv0[r], sv1[r]));
        rm = fmaxf(rm, __shfl_xor(rm, 16));
        const float mnew = fmaxf(mrun, rm);
        const float corr = __expf(mrun - mnew);

        // P fragment: elements 0..7 <- s0 tile, 8..15 <- s1 tile, packed pairs
        Frag p;
        float rs = 0.0f;
#pragma unroll
        for (int q = 0; q < 4; q++) {
            float e0 = __expf(sv0[2 * q]     - mnew);
            float e1 = __expf(sv0[2 * q + 1] - mnew);
            float e2 = __expf(sv1[2 * q]     - mnew);
            float e3 = __expf(sv1[2 * q + 1] - mnew);
            p.pk[q]     = __builtin_amdgcn_cvt_pkrtz(e0, e1);
            p.pk[q + 4] = __builtin_amdgcn_cvt_pkrtz(e2, e3);
            rs += (e0 + e1) + (e2 + e3);
        }
        rs += __shfl_xor(rs, 16);
        lrun = lrun * corr + rs;
        mrun = mnew;

        // rescale O accumulators (per-row correction broadcast)
#pragma unroll
        for (int r = 0; r < 8; r++) {
            const float cr = __shfl(corr, r + hi * 8);
#pragma unroll
            for (int nt = 0; nt < 4; nt++) o[nt][r] *= cr;
        }

        // V B-fragments from V^T: lane = column d, j chunks at +0 and +16
#pragma unroll
        for (int nt = 0; nt < 4; nt++) {
            Frag vb;
            const _Float16* pv =
                Vh + (size_t)(nt * 16 + ln) * SEQT + kb + hi * 8;
            vb.h[0] = *(const v8h*)pv;
            vb.h[1] = *(const v8h*)(pv + 16);
            o[nt] = wmma16(p.v, vb.v, o[nt]);
        }
    }

    // normalize by running sum and store x[b, t, h*64 + d] (f16)
    const float linv = 1.0f / lrun;
#pragma unroll
    for (int r = 0; r < 8; r++) {
        const float lr = __shfl(linv, r + hi * 8);
        const int t = q0 + r + hi * 8;
        _Float16* xr = X + (size_t)(bb * SEQT + t) * D_MODEL + hd * DKH;
#pragma unroll
        for (int nt = 0; nt < 4; nt++)
            xr[nt * 16 + ln] = (_Float16)(o[nt][r] * lr);
    }
}

// ---------------------------------------------------------------------------
extern "C" void kernel_launch(void* const* d_in, const int* in_sizes, int n_in,
                              void* d_out, int out_size, void* d_ws, size_t ws_size,
                              hipStream_t stream) {
    (void)in_sizes; (void)n_in; (void)out_size; (void)ws_size;

    const float* Q    = (const float*)d_in[0];
    const float* K    = (const float*)d_in[1];
    const float* V    = (const float*)d_in[2];
    const int*   mask = (const int*)d_in[3];
    const float* WQw  = (const float*)d_in[4];
    const float* WQb  = (const float*)d_in[5];
    const float* WKw  = (const float*)d_in[6];
    const float* WKb  = (const float*)d_in[7];
    const float* WVw  = (const float*)d_in[8];
    const float* WVb  = (const float*)d_in[9];
    const float* WOw  = (const float*)d_in[10];
    const float* WOb  = (const float*)d_in[11];
    float* out = (float*)d_out;

    const size_t NQ = (size_t)MROWS * D_MODEL;    // 8,388,608
    const size_t NW = (size_t)D_MODEL * D_MODEL;  // 1,048,576

    _Float16* ws  = (_Float16*)d_ws;
    _Float16* hQ  = ws;
    _Float16* hK  = hQ + NQ;
    _Float16* hV  = hK + NQ;
    _Float16* hWq = hV + NQ;
    _Float16* hWk = hWq + NW;
    _Float16* hWv = hWk + NW;
    _Float16* hWo = hWv + NW;
    _Float16* qp  = hWo + NW;   // f16 [B*T, D_MODEL]
    _Float16* kp  = qp + NQ;    // f16 [B*T, D_MODEL]
    _Float16* vt  = kp + NQ;    // f16 V^T [B, H, DK, T]
    _Float16* x   = vt + NQ;    // f16 [B*T, D_MODEL]

    // Stage 0: convert inputs + weights to f16
    cvt_f32_f16<<<(int)(NQ / 2048), 256, 0, stream>>>(Q, hQ, (int)NQ);
    cvt_f32_f16<<<(int)(NQ / 2048), 256, 0, stream>>>(K, hK, (int)NQ);
    cvt_f32_f16<<<(int)(NQ / 2048), 256, 0, stream>>>(V, hV, (int)NQ);
    cvt_f32_f16<<<(int)(NW / 2048), 256, 0, stream>>>(WQw, hWq, (int)NW);
    cvt_f32_f16<<<(int)(NW / 2048), 256, 0, stream>>>(WKw, hWk, (int)NW);
    cvt_f32_f16<<<(int)(NW / 2048), 256, 0, stream>>>(WVw, hWv, (int)NW);
    cvt_f32_f16<<<(int)(NW / 2048), 256, 0, stream>>>(WOw, hWo, (int)NW);

    // Stage 1: QKV projections (Q/K row-major f16, V transposed head-split)
    const int gemmGrid = (MROWS / 128) * (D_MODEL / 256);  // 256 blocks
    gemm_wmma<0><<<gemmGrid, 256, 0, stream>>>(hQ, hWq, WQb, qp, nullptr,
                                               MROWS, D_MODEL, D_MODEL);
    gemm_wmma<0><<<gemmGrid, 256, 0, stream>>>(hK, hWk, WKb, kp, nullptr,
                                               MROWS, D_MODEL, D_MODEL);
    gemm_wmma<2><<<gemmGrid, 256, 0, stream>>>(hV, hWv, WVb, vt, nullptr,
                                               MROWS, D_MODEL, D_MODEL);

    // Stage 2: attention (8192 waves, 8 per block)
    attn_kernel<<<(BATCH * NH * (SEQT / 16)) / 8, 256, 0, stream>>>(
        qp, kp, vt, mask, x);

    // Stage 3: output projection -> f32 d_out
    gemm_wmma<1><<<gemmGrid, 256, 0, stream>>>(x, hWo, WOb, nullptr, out,
                                               MROWS, D_MODEL, D_MODEL);
}
